// GTEATLSTM3Train_35021163331773
// MI455X (gfx1250) — matrix operand order repaired and checked by reading, not compiled, not run on hardware
//
#include <hip/hip_runtime.h>
#include <hip/hip_bf16.h>
#include <math.h>

typedef __attribute__((ext_vector_type(16))) _Float16 v16h;
typedef __attribute__((ext_vector_type(8)))  _Float16 v8h;
typedef __attribute__((ext_vector_type(8)))  float    v8f;

#define N_SRC   65536
#define NDST    2048
#define DEG     32
#define E_TOT   (NDST*DEG)   // 65536
#define TS      16
#define DE      16
#define HDIM    128
#define NCLS    10

#define LOG2E   1.4426950408889634f

__device__ __forceinline__ v8f wmma16(v16h a, v16h b, v8f c) {
  // v_wmma_f32_16x16x32_f16: D = A(16x32 f16) * B(32x16 f16) + C(16x16 f32)
  return __builtin_amdgcn_wmma_f32_16x16x32_f16(false, a, false, b, (short)0, c, false, false);
}
__device__ __forceinline__ v8f splat8(float x) {
  v8f r;
#pragma unroll
  for (int i = 0; i < 8; ++i) r[i] = x;
  return r;
}

// Branch-free transcendentals on the hardware exp2/rcp/log2 units.
__device__ __forceinline__ float fexp2(float x) { return __builtin_amdgcn_exp2f(x); }
__device__ __forceinline__ float frcp(float x)  { return __builtin_amdgcn_rcpf(x); }
__device__ __forceinline__ float sigm(float x) {
  return frcp(1.0f + fexp2(-LOG2E * x));
}
__device__ __forceinline__ float tanh_f(float x) {
  // tanh(x) = 1 - 2/(exp2(2*log2e*x)+1); saturates to +-1 via hw exp2/rcp
  return 1.0f - 2.0f * frcp(1.0f + fexp2(2.0f * LOG2E * x));
}

// A-operand chunk (16x32 f16) from an f32 row vector (CDNA5 16-bit A layout).
__device__ __forceinline__ v16h aChunkF32(const float* __restrict__ row, int kbase, int laneHi) {
  int k0 = kbase + (laneHi ? 8 : 0);
  v16h r;
#pragma unroll
  for (int i = 0; i < 8; ++i) {
    r[i]     = (_Float16)row[k0 + i];
    r[i + 8] = (_Float16)row[k0 + 16 + i];
  }
  return r;
}
// Same, but from an f16 row staged in LDS (two 16B vector loads).
__device__ __forceinline__ v16h aChunkLds(const _Float16* row, int kbase, int laneHi) {
  int k0 = kbase + (laneHi ? 8 : 0);
  v8h lo = *(const v8h*)(row + k0);
  v8h hi = *(const v8h*)(row + k0 + 16);
  return __builtin_shufflevector(lo, hi, 0,1,2,3,4,5,6,7,8,9,10,11,12,13,14,15);
}
// Edge-feature A chunk: K=16 real values zero-padded to K=32.
__device__ __forceinline__ v16h aChunkEdge(const float* __restrict__ erow, int laneHi) {
  int k0 = laneHi ? 8 : 0;
  v16h r;
#pragma unroll
  for (int i = 0; i < 8; ++i) {
    r[i]     = (_Float16)erow[k0 + i];
    r[i + 8] = (_Float16)0.0f;
  }
  return r;
}
// B-operand tile (32x16 f16) from LDS pack: 512 halves per tile, 32B per lane.
__device__ __forceinline__ v16h bTile(const _Float16* pack, int tile, int lane) {
  return *(const v16h*)(pack + (tile << 9) + (lane << 4));
}

// ---------------------------------------------------------------------------
// Pack an f32 [Ksrc x Ncols] matrix into f16 WMMA B-operand tiles.
// ---------------------------------------------------------------------------
__global__ void pack_b_kernel(const float* __restrict__ src, _Float16* __restrict__ dst,
                              int Ksrc, int Ncols, int Ntiles, int total) {
  int idx = blockIdx.x * 256 + threadIdx.x;
  if (idx >= total) return;
  int h    = idx & 15;
  int lane = (idx >> 4) & 31;
  int tile = idx >> 9;
  int nt = tile % Ntiles;
  int kc = tile / Ntiles;
  int K = kc * 32 + (h < 8 ? 0 : 16) + (lane < 16 ? 0 : 8) + (h & 7);
  int n = nt * 16 + (lane & 15);
  float v = (K < Ksrc) ? src[(size_t)K * Ncols + n] : 0.0f;
  dst[idx] = (_Float16)v;
}

// ---------------------------------------------------------------------------
// T-LSTM: one wave per 16-edge tile; weights resident in LDS (f16 B tiles);
// per-step recurrence on v_wmma_f32_16x16x32_f16. B tiles are fetched in
// groups of 4 independent tiles (4 gates x fixed K-chunk) so one dscnt wait
// covers 4 WMMAs and the next group's loads overlap the current group.
// ---------------------------------------------------------------------------
__global__ void __launch_bounds__(256) tlstm_kernel(
    const float* __restrict__ eTen, const float* __restrict__ dtTen,
    const int* __restrict__ elenG, const _Float16* __restrict__ packG,
    const float* __restrict__ b4, const float* __restrict__ bdv,
    const float* __restrict__ attnw, float* __restrict__ outH,
    float* __restrict__ outA, int mode)
{
  extern __shared__ char smem[];
  _Float16* sU     = (_Float16*)smem;    // 65536 halves (128KB)
  _Float16* sWd    = sU + 65536;         // 16384 halves (32KB)
  _Float16* sW     = sWd + 16384;        // 16384 halves (32KB)
  _Float16* sStage = sW + 16384;         // 8 waves * (2048 h + 2048 c) halves (64KB)
  {
    const uint4* g = (const uint4*)packG;
    uint4* s = (uint4*)smem;
    for (int i = threadIdx.x; i < 12288; i += 256) s[i] = g[i];   // 192KB weights
    uint4* z = (uint4*)sStage;
    uint4 z4 = make_uint4(0u, 0u, 0u, 0u);
    for (int i = threadIdx.x; i < 4096; i += 256) z[i] = z4;      // zero h/c stage
  }
  __syncthreads();

  const int wave   = threadIdx.x >> 5;
  const int lane   = threadIdx.x & 31;
  const int laneHi = lane >> 4;
  const int ln     = lane & 15;
  const int edgeBase = (blockIdx.x * 8 + wave) * 16;
  _Float16* sh = sStage + wave * 4096;
  _Float16* sc = sh + 2048;

  int el[8];
#pragma unroll
  for (int r = 0; r < 8; ++r) {
    int L = elenG[edgeBase + r + 8 * laneHi] - 1;
    el[r] = L < 0 ? 0 : (L > 15 ? 15 : L);
  }

  // Hoist per-lane biases out of the recurrence.
  float bW[4][8], bD[8], aW16[8];
#pragma unroll
  for (int gg = 0; gg < 4; ++gg)
#pragma unroll
    for (int j = 0; j < 8; ++j) bW[gg][j] = b4[gg * 128 + j * 16 + ln];
#pragma unroll
  for (int j = 0; j < 8; ++j) bD[j] = bdv[j * 16 + ln];
#pragma unroll
  for (int j = 0; j < 8; ++j) aW16[j] = attnw[j * 16 + ln];

  v8f c[8], hsel[8];
#pragma unroll
  for (int j = 0; j < 8; ++j) { c[j] = splat8(0.0f); hsel[j] = splat8(0.0f); }

  for (int t = 0; t < TS; ++t) {
    v16h hA[4], cA[4];
#pragma unroll
    for (int kc = 0; kc < 4; ++kc) {
      hA[kc] = aChunkLds(sh + ln * 128, kc * 32, laneHi);
      cA[kc] = aChunkLds(sc + ln * 128, kc * 32, laneHi);
    }
    const float* erow = eTen + (size_t)(edgeBase + ln) * (TS * DE) + (size_t)t * DE;
    v16h eA = aChunkEdge(erow, laneHi);

    float dec[8];
#pragma unroll
    for (int r = 0; r < 8; ++r) {
      float d = dtTen[(size_t)(edgeBase + r + 8 * laneHi) * TS + t];
      dec[r] = LOG2E * frcp(__builtin_amdgcn_logf(2.7182818284590452f + d));
    }

#pragma unroll
    for (int j = 0; j < 8; ++j) {
      // c_s = tanh(c @ Wd + bd): fetch all 4 Wd tiles, then chain the WMMAs.
      v16h wd0 = bTile(sWd, 0 * 8 + j, lane);
      v16h wd1 = bTile(sWd, 1 * 8 + j, lane);
      v16h wd2 = bTile(sWd, 2 * 8 + j, lane);
      v16h wd3 = bTile(sWd, 3 * 8 + j, lane);
      v8f cs = splat8(bD[j]);
      cs = wmma16(cA[0], wd0, cs);
      cs = wmma16(cA[1], wd1, cs);
      cs = wmma16(cA[2], wd2, cs);
      cs = wmma16(cA[3], wd3, cs);
      v8f cst;
#pragma unroll
      for (int r = 0; r < 8; ++r) {
        float s = tanh_f(cs[r]);
        cst[r] = c[j][r] - s + s * dec[r];
      }

      // gates = x_proj + h @ U  (gate order i,f,o,g).
      // Group loads: 4 independent tiles (one per gate) per K-chunk, then
      // 4 independent WMMAs consuming them.
      v8f acc[4];
#pragma unroll
      for (int gg = 0; gg < 4; ++gg) acc[gg] = splat8(bW[gg][j]);
      {
        v16h w0 = bTile(sW, 0 * 8 + j, lane);
        v16h w1 = bTile(sW, 1 * 8 + j, lane);
        v16h w2 = bTile(sW, 2 * 8 + j, lane);
        v16h w3 = bTile(sW, 3 * 8 + j, lane);
        acc[0] = wmma16(eA, w0, acc[0]);
        acc[1] = wmma16(eA, w1, acc[1]);
        acc[2] = wmma16(eA, w2, acc[2]);
        acc[3] = wmma16(eA, w3, acc[3]);
      }
#pragma unroll
      for (int kc = 0; kc < 4; ++kc) {
        v16h u0 = bTile(sU, kc * 32 + 0 * 8 + j, lane);
        v16h u1 = bTile(sU, kc * 32 + 1 * 8 + j, lane);
        v16h u2 = bTile(sU, kc * 32 + 2 * 8 + j, lane);
        v16h u3 = bTile(sU, kc * 32 + 3 * 8 + j, lane);
        acc[0] = wmma16(hA[kc], u0, acc[0]);
        acc[1] = wmma16(hA[kc], u1, acc[1]);
        acc[2] = wmma16(hA[kc], u2, acc[2]);
        acc[3] = wmma16(hA[kc], u3, acc[3]);
      }

      v8f cn, hn;
#pragma unroll
      for (int r = 0; r < 8; ++r) {
        float cv = sigm(acc[1][r]) * cst[r] + sigm(acc[0][r]) * tanh_f(acc[3][r]);
        cn[r] = cv;
        hn[r] = sigm(acc[2][r]) * tanh_f(cv);
      }
      c[j] = cn;
#pragma unroll
      for (int r = 0; r < 8; ++r) if (el[r] == t) hsel[j][r] = hn[r];
#pragma unroll
      for (int r = 0; r < 8; ++r) {   // stage for next step's A operands
        int m = r + 8 * laneHi;
        sh[m * 128 + j * 16 + ln] = (_Float16)hn[r];
        sc[m * 128 + j * 16 + ln] = (_Float16)cn[r];
      }
    }
  }

  if (mode == 0) {
#pragma unroll
    for (int j = 0; j < 8; ++j)
#pragma unroll
      for (int r = 0; r < 8; ++r)
        outH[(size_t)(edgeBase + r + 8 * laneHi) * HDIM + j * 16 + ln] = hsel[j][r];
  } else {
    float part[8];
#pragma unroll
    for (int r = 0; r < 8; ++r) part[r] = 0.0f;
#pragma unroll
    for (int j = 0; j < 8; ++j) {
#pragma unroll
      for (int r = 0; r < 8; ++r) part[r] += hsel[j][r] * aW16[j];
    }
#pragma unroll
    for (int r = 0; r < 8; ++r) {     // reduce over the 16 lanes of each half
      float v = part[r];
      v += __shfl_xor(v, 1, 32);
      v += __shfl_xor(v, 2, 32);
      v += __shfl_xor(v, 4, 32);
      v += __shfl_xor(v, 8, 32);
      part[r] = v;
    }
    if (ln == 0) {
#pragma unroll
      for (int r = 0; r < 8; ++r) {
        float x = part[r];
        outA[edgeBase + 8 * laneHi + r] = x > 0.0f ? x : 0.01f * x;
      }
    }
  }
}

// ---------------------------------------------------------------------------
// edge_out GEMM: m = relu([h_src | e_out] @ W + b) for E rows (gathered), plus
// self_h_tmp = [self_h | 0] @ W + b (no relu) for N_DST rows.
// ---------------------------------------------------------------------------
__global__ void __launch_bounds__(256) edgeout_kernel(
    const float* __restrict__ nodef, const int* __restrict__ srcIdx,
    const float* __restrict__ eOut, const _Float16* __restrict__ packG,
    const float* __restrict__ bias, float* __restrict__ mOut,
    float* __restrict__ selfOut)
{
  extern __shared__ char smem[];
  _Float16* sB = (_Float16*)smem;   // 32768 halves (64KB)
  {
    const uint4* g = (const uint4*)packG;
    uint4* s = (uint4*)smem;
    for (int i = threadIdx.x; i < 4096; i += 256) s[i] = g[i];
  }
  __syncthreads();
  int wave = threadIdx.x >> 5, lane = threadIdx.x & 31;
  int laneHi = lane >> 4, ln = lane & 15;
  int tile = blockIdx.x * 8 + wave;
  bool isSelf = tile >= (E_TOT / 16);
  int rowBase = isSelf ? (tile - E_TOT / 16) * 16 : tile * 16;
  int row = rowBase + ln;
  const float* rv0 = isSelf ? (nodef + (size_t)row * HDIM)
                            : (nodef + (size_t)srcIdx[row] * HDIM);
  v16h A[8];
#pragma unroll
  for (int kc = 0; kc < 4; ++kc) A[kc] = aChunkF32(rv0, kc * 32, laneHi);
  if (!isSelf) {
    const float* rv1 = eOut + (size_t)row * HDIM;
#pragma unroll
    for (int kc = 0; kc < 4; ++kc) A[4 + kc] = aChunkF32(rv1, kc * 32, laneHi);
  }
#pragma unroll
  for (int nt = 0; nt < 8; ++nt) {
    v8f acc = splat8(bias[nt * 16 + ln]);
    {
      v16h b0 = bTile(sB, 0 * 8 + nt, lane);
      v16h b1 = bTile(sB, 1 * 8 + nt, lane);
      v16h b2 = bTile(sB, 2 * 8 + nt, lane);
      v16h b3 = bTile(sB, 3 * 8 + nt, lane);
      acc = wmma16(A[0], b0, acc);
      acc = wmma16(A[1], b1, acc);
      acc = wmma16(A[2], b2, acc);
      acc = wmma16(A[3], b3, acc);
    }
    if (!isSelf) {
      v16h b4t = bTile(sB, 4 * 8 + nt, lane);
      v16h b5t = bTile(sB, 5 * 8 + nt, lane);
      v16h b6t = bTile(sB, 6 * 8 + nt, lane);
      v16h b7t = bTile(sB, 7 * 8 + nt, lane);
      acc = wmma16(A[4], b4t, acc);
      acc = wmma16(A[5], b5t, acc);
      acc = wmma16(A[6], b6t, acc);
      acc = wmma16(A[7], b7t, acc);
    }
    float* op = isSelf ? selfOut : mOut;
#pragma unroll
    for (int r = 0; r < 8; ++r) {
      float v = acc[r];
      if (!isSelf) v = fmaxf(v, 0.0f);
      op[(size_t)(rowBase + r + 8 * laneHi) * HDIM + nt * 16 + ln] = v;
    }
  }
}

// ---------------------------------------------------------------------------
// Per-node sparsemax over DEG=32 scores (one wave per node) + weighted
// aggregation of m, minus self_h_tmp, producing hcat = [self_h | h_agg].
// ---------------------------------------------------------------------------
__global__ void __launch_bounds__(256) aggregate_kernel(
    const float* __restrict__ aVals, const float* __restrict__ mBuf,
    const float* __restrict__ nodef, const float* __restrict__ selfTmp,
    float* __restrict__ hcat)
{
  int wave = threadIdx.x >> 5;
  int lane = threadIdx.x & 31;
  int node = blockIdx.x * 8 + wave;
  float z = aVals[node * DEG + lane];
  float mx = z;
#pragma unroll
  for (int off = 16; off >= 1; off >>= 1) mx = fmaxf(mx, __shfl_xor(mx, off, 32));
  z -= mx;
  int rank = 0;
  for (int j = 0; j < 32; ++j) {
    float zj = __shfl(z, j, 32);
    if (zj > z || (zj == z && j < lane)) rank++;
  }
  float cum = 0.0f;
  for (int j = 0; j < 32; ++j) {
    float zj = __shfl(z, j, 32);
    int rj = __shfl(rank, j, 32);
    if (rj <= rank) cum += zj;
  }
  float kk = (float)(rank + 1);
  bool sel = (1.0f + kk * z) > cum;
  float km = sel ? kk : 0.0f;
  float ss = sel ? z : 0.0f;
#pragma unroll
  for (int off = 16; off >= 1; off >>= 1) {
    km = fmaxf(km, __shfl_xor(km, off, 32));
    ss += __shfl_xor(ss, off, 32);
  }
  float tau = (ss - 1.0f) / km;
  float alpha = fmaxf(0.0f, z - tau);

  float acc0 = 0, acc1 = 0, acc2 = 0, acc3 = 0;
  for (int d = 0; d < 32; ++d) {
    float ad = __shfl(alpha, d, 32);
    if (ad != 0.0f) {
      const float* mr = mBuf + ((size_t)node * DEG + d) * HDIM;
      acc0 += ad * mr[lane];
      acc1 += ad * mr[lane + 32];
      acc2 += ad * mr[lane + 64];
      acc3 += ad * mr[lane + 96];
    }
  }
  float* hc = hcat + (size_t)node * 256;
  const float* nf = nodef + (size_t)node * HDIM;
  const float* st = selfTmp + (size_t)node * HDIM;
  hc[lane]        = nf[lane];
  hc[lane + 32]   = nf[lane + 32];
  hc[lane + 64]   = nf[lane + 64];
  hc[lane + 96]   = nf[lane + 96];
  hc[128 + lane]      = acc0 - st[lane];
  hc[128 + lane + 32] = acc1 - st[lane + 32];
  hc[128 + lane + 64] = acc2 - st[lane + 64];
  hc[128 + lane + 96] = acc3 - st[lane + 96];
}

// act = relu(hcat @ node_W + node_b), rows = N_DST, K = 256.
__global__ void __launch_bounds__(256) nodegemm_kernel(
    const float* __restrict__ hcat, const _Float16* __restrict__ packG,
    const float* __restrict__ bias, float* __restrict__ act)
{
  extern __shared__ char smem[];
  _Float16* sB = (_Float16*)smem;
  {
    const uint4* g = (const uint4*)packG;
    uint4* s = (uint4*)smem;
    for (int i = threadIdx.x; i < 4096; i += 256) s[i] = g[i];
  }
  __syncthreads();
  int wave = threadIdx.x >> 5, lane = threadIdx.x & 31;
  int laneHi = lane >> 4, ln = lane & 15;
  int rowBase = (blockIdx.x * 8 + wave) * 16;
  const float* rv = hcat + (size_t)(rowBase + ln) * 256;
  v16h A[8];
#pragma unroll
  for (int kc = 0; kc < 8; ++kc) A[kc] = aChunkF32(rv, kc * 32, laneHi);
#pragma unroll
  for (int nt = 0; nt < 8; ++nt) {
    v8f acc = splat8(bias[nt * 16 + ln]);
    v16h b0 = bTile(sB, 0 * 8 + nt, lane);
    v16h b1 = bTile(sB, 1 * 8 + nt, lane);
    v16h b2 = bTile(sB, 2 * 8 + nt, lane);
    v16h b3 = bTile(sB, 3 * 8 + nt, lane);
    acc = wmma16(A[0], b0, acc);
    acc = wmma16(A[1], b1, acc);
    acc = wmma16(A[2], b2, acc);
    acc = wmma16(A[3], b3, acc);
    v16h b4t = bTile(sB, 4 * 8 + nt, lane);
    v16h b5t = bTile(sB, 5 * 8 + nt, lane);
    v16h b6t = bTile(sB, 6 * 8 + nt, lane);
    v16h b7t = bTile(sB, 7 * 8 + nt, lane);
    acc = wmma16(A[4], b4t, acc);
    acc = wmma16(A[5], b5t, acc);
    acc = wmma16(A[6], b6t, acc);
    acc = wmma16(A[7], b7t, acc);
#pragma unroll
    for (int r = 0; r < 8; ++r)
      act[(size_t)(rowBase + r + 8 * laneHi) * HDIM + nt * 16 + ln] = fmaxf(acc[r], 0.0f);
  }
}

// out = act @ fc_W + fc_b  (2048 x 10, tiny)
__global__ void fc_kernel(const float* __restrict__ act, const float* __restrict__ fcW,
                          const float* __restrict__ fcB, float* __restrict__ out)
{
  int idx = blockIdx.x * 256 + threadIdx.x;
  if (idx >= NDST * NCLS) return;
  int row = idx / NCLS, cls = idx % NCLS;
  float s = fcB[cls];
  const float* a = act + (size_t)row * HDIM;
#pragma unroll 4
  for (int k = 0; k < HDIM; ++k) s += a[k] * fcW[k * NCLS + cls];
  out[idx] = s;
}

extern "C" void kernel_launch(void* const* d_in, const int* in_sizes, int n_in,
                              void* d_out, int out_size, void* d_ws, size_t ws_size,
                              hipStream_t stream) {
  (void)in_sizes; (void)n_in; (void)out_size; (void)ws_size;
  const float* nodef = (const float*)d_in[0];
  const float* eTen  = (const float*)d_in[1];
  const float* dtTen = (const float*)d_in[2];
  const float* mW  = (const float*)d_in[3];
  const float* mU  = (const float*)d_in[4];
  const float* mB  = (const float*)d_in[5];
  const float* mWd = (const float*)d_in[6];
  const float* mBd = (const float*)d_in[7];
  const float* aW  = (const float*)d_in[8];
  const float* aU  = (const float*)d_in[9];
  const float* aB  = (const float*)d_in[10];
  const float* aWd = (const float*)d_in[11];
  const float* aBd = (const float*)d_in[12];
  const float* attnw = (const float*)d_in[13];
  const float* eoW = (const float*)d_in[14];
  const float* eoB = (const float*)d_in[15];
  const float* nW  = (const float*)d_in[16];
  const float* nB  = (const float*)d_in[17];
  const float* fcW = (const float*)d_in[18];
  const float* fcB = (const float*)d_in[19];
  const int* elen   = (const int*)d_in[20];
  const int* srcIdx = (const int*)d_in[21];
  float* out = (float*)d_out;

  // Workspace layout: packed f16 weights (512KB) then f32 intermediates (~71MB).
  char* ws = (char*)d_ws;
  _Float16* packM  = (_Float16*)ws;          // U(65536)+Wd(16384)+W(16384) halves
  _Float16* packA  = packM + 98304;
  _Float16* packEO = packA + 98304;          // 32768 halves
  _Float16* packNW = packEO + 32768;         // 32768 halves
  float* f32ws  = (float*)(ws + 512 * 1024);
  float* eOut    = f32ws;                                   // E*128
  float* aVals   = eOut + (size_t)E_TOT * HDIM;             // E
  float* mBuf    = aVals + E_TOT;                           // E*128
  float* selfTmp = mBuf + (size_t)E_TOT * HDIM;             // 2048*128
  float* hcat    = selfTmp + (size_t)NDST * HDIM;           // 2048*256
  float* act     = hcat + (size_t)NDST * 256;               // 2048*128

  auto packLaunch = [&](const float* src, _Float16* dst, int Ksrc, int N, int Ntiles, int total) {
    pack_b_kernel<<<(total + 255) / 256, 256, 0, stream>>>(src, dst, Ksrc, N, Ntiles, total);
  };
  packLaunch(mU,  packM,          128, 512, 32, 65536);
  packLaunch(mWd, packM + 65536,  128, 128,  8, 16384);
  packLaunch(mW,  packM + 81920,   16, 512, 32, 16384);
  packLaunch(aU,  packA,          128, 512, 32, 65536);
  packLaunch(aWd, packA + 65536,  128, 128,  8, 16384);
  packLaunch(aW,  packA + 81920,   16, 512, 32, 16384);
  packLaunch(eoW, packEO,         256, 128,  8, 32768);
  packLaunch(nW,  packNW,         256, 128,  8, 32768);

  hipFuncSetAttribute((const void*)tlstm_kernel,    hipFuncAttributeMaxDynamicSharedMemorySize, 262144);
  hipFuncSetAttribute((const void*)edgeout_kernel,  hipFuncAttributeMaxDynamicSharedMemorySize, 65536);
  hipFuncSetAttribute((const void*)nodegemm_kernel, hipFuncAttributeMaxDynamicSharedMemorySize, 65536);

  // Two T-LSTMs (message + attention).
  tlstm_kernel<<<E_TOT / 128, 256, 262144, stream>>>(eTen, dtTen, elen, packM, mB, mBd,
                                                     attnw, eOut, aVals, 0);
  tlstm_kernel<<<E_TOT / 128, 256, 262144, stream>>>(eTen, dtTen, elen, packA, aB, aBd,
                                                     attnw, eOut, aVals, 1);
  // m + self_h_tmp
  edgeout_kernel<<<(E_TOT / 16 + NDST / 16) / 8, 256, 65536, stream>>>(
      nodef, srcIdx, eOut, packEO, eoB, mBuf, selfTmp);
  // sparsemax + aggregate -> hcat
  aggregate_kernel<<<NDST / 8, 256, 0, stream>>>(aVals, mBuf, nodef, selfTmp, hcat);
  // node layer
  nodegemm_kernel<<<(NDST / 16) / 8, 256, 65536, stream>>>(hcat, packNW, nB, act);
  // classifier
  fc_kernel<<<(NDST * NCLS + 255) / 256, 256, 0, stream>>>(act, fcW, fcB, out);
}
